// TemporalMamba_91225105367675
// MI455X (gfx1250) — matrix operand order repaired and checked by reading, not compiled, not run on hardware
//
#include <hip/hip_runtime.h>

typedef _Float16 f16;
typedef __attribute__((ext_vector_type(16))) _Float16 v16h;
typedef __attribute__((ext_vector_type(8)))  _Float16 v8h;
typedef __attribute__((ext_vector_type(8)))  float    v8f;
typedef __attribute__((ext_vector_type(4)))  int      i32x4;

#ifndef __has_builtin
#define __has_builtin(x) 0
#endif

#if __has_builtin(__builtin_amdgcn_global_load_async_to_lds_b128)
#define HAVE_ASYNC_LDS 1
#else
#define HAVE_ASYNC_LDS 0
#endif

// 16-byte global -> LDS copy: async engine when available (ASYNCcnt path,
// no VGPR round trip), otherwise load+store through registers.
__device__ __forceinline__ void cp16_g2l(const f16* __restrict__ g, f16* __restrict__ l) {
#if HAVE_ASYNC_LDS
  __builtin_amdgcn_global_load_async_to_lds_b128(
      (__attribute__((address_space(1))) i32x4*)g,
      (__attribute__((address_space(3))) i32x4*)l, 0, 0);
#else
  *(v8h*)l = *(const v8h*)g;
#endif
}

__device__ __forceinline__ void wait_async_copies() {
#if HAVE_ASYNC_LDS
#if __has_builtin(__builtin_amdgcn_s_wait_asynccnt)
  __builtin_amdgcn_s_wait_asynccnt(0);
#else
  asm volatile("s_wait_asynccnt 0" ::: "memory");
#endif
#endif
}

#define BATCH   4
#define HH      64
#define WW      64
#define LL      4096
#define MROWS   (BATCH*LL)     // 16384
#define DIN     512            // DINNER
#define DST     128            // DSTATE
#define CVD     768            // CONVDIM
#define DPJ     1288           // DINPROJ
#define NHD     8

#define SEG     64             // scan segments
#define TS      64             // steps per segment

#define BM      128
#define BN      64
#define BK      64

// ---------------------------------------------------------------------------
// elementwise helpers
// ---------------------------------------------------------------------------
__global__ void f2h_kernel(const float* __restrict__ s, f16* __restrict__ d, int n) {
  int i = blockIdx.x * blockDim.x + threadIdx.x;
  if (i < n) d[i] = (f16)s[i];
}

// conv3_w (128,512,3,3) -> f16 [o=ky*3+kx][oc][ic]
__global__ void conv3_repack_kernel(const float* __restrict__ w, f16* __restrict__ d) {
  int i = blockIdx.x * blockDim.x + threadIdx.x;
  if (i >= 9 * 128 * 512) return;
  int o = i / (128 * 512);
  int r = i - o * (128 * 512);
  int oc = r >> 9, ic = r & 511;
  d[i] = (f16)w[(size_t)(oc * 512 + ic) * 9 + o];
}

__global__ void build_cat_kernel(const float* __restrict__ curr, const float* __restrict__ prev,
                                 f16* __restrict__ cat) {
  int i = blockIdx.x * blockDim.x + threadIdx.x;
  if (i >= MROWS * 512) return;
  int c = i & 511, m = i >> 9;
  float v = (c < 256) ? curr[(size_t)m * 256 + c] : prev[(size_t)m * 256 + (c - 256)];
  cat[i] = (f16)v;
}

// ---------------------------------------------------------------------------
// WMMA GEMM: C[M,N] = A[M,K] * W[N,K]^T (+bias)(+resid); f32 and/or f16 out
// 256 threads = 8 waves (4Mx2N), block tile 128x64, K-stage 64 ->
// 8 WMMAs per wave per barrier pair. Tiles staged via async global->LDS.
// ---------------------------------------------------------------------------
__global__ __launch_bounds__(256)
void gemm_wmma_kernel(const f16* __restrict__ A, const f16* __restrict__ W,
                      const float* __restrict__ bias, const float* __restrict__ resid,
                      float* __restrict__ C, f16* __restrict__ Ch,
                      int M, int N, int K) {
  __shared__ alignas(16) f16 lA[BM][BK];
  __shared__ alignas(16) f16 lB[BN][BK];
  int tid  = threadIdx.x;
  int lane = tid & 31, wave = tid >> 5;
  int wm = wave >> 1, wn = wave & 1;
  int bm = blockIdx.x * BM;
  int bn = blockIdx.y * BN;
  int r  = lane & 15, hi = lane >> 4;
  int k1 = hi ? 8 : 0, k2 = hi ? 24 : 16;

  float bv[2];
#pragma unroll
  for (int ni = 0; ni < 2; ni++) {
    int ng = bn + wn * 32 + ni * 16 + r;
    bv[ni] = (bias && ng < N) ? bias[ng] : 0.0f;
  }
  v8f acc[2][2];
#pragma unroll
  for (int mi = 0; mi < 2; mi++)
#pragma unroll
    for (int ni = 0; ni < 2; ni++)
#pragma unroll
      for (int i = 0; i < 8; i++) acc[mi][ni][i] = bv[ni];

  for (int kb = 0; kb < K; kb += BK) {
    __syncthreads();
#pragma unroll
    for (int q = 0; q < 4; q++) {            // stage A: 128x64 halves
      int ci = q * 256 + tid;
      int row = ci >> 3, col = (ci & 7) * 8;
      cp16_g2l(&A[(size_t)(bm + row) * K + kb + col], &lA[row][col]);
      if (kb + BK < K)
        __builtin_prefetch(&A[(size_t)(bm + row) * K + kb + BK + col], 0, 0);
    }
#pragma unroll
    for (int q = 0; q < 2; q++) {            // stage B: 64x64 halves
      int ci = q * 256 + tid;
      int row = ci >> 3, col = (ci & 7) * 8;
      int nrow = bn + row;
      if (nrow < N) {
        cp16_g2l(&W[(size_t)nrow * K + kb + col], &lB[row][col]);
      } else {
        v8h wv;
#pragma unroll
        for (int i = 0; i < 8; i++) wv[i] = (f16)0.0f;
        *(v8h*)&lB[row][col] = wv;
      }
    }
    wait_async_copies();
    __syncthreads();
#pragma unroll
    for (int ks = 0; ks < BK; ks += 32) {
      v16h a[2], b[2];
#pragma unroll
      for (int mi = 0; mi < 2; mi++) {
        int row = wm * 32 + mi * 16 + r;
        v8h alo = *(const v8h*)&lA[row][ks + k1];
        v8h ahi = *(const v8h*)&lA[row][ks + k2];
#pragma unroll
        for (int i = 0; i < 8; i++) { a[mi][i] = alo[i]; a[mi][i + 8] = ahi[i]; }
      }
#pragma unroll
      for (int ni = 0; ni < 2; ni++) {
        int row = wn * 32 + ni * 16 + r;
        v8h blo = *(const v8h*)&lB[row][ks + k1];
        v8h bhi = *(const v8h*)&lB[row][ks + k2];
#pragma unroll
        for (int i = 0; i < 8; i++) { b[ni][i] = blo[i]; b[ni][i + 8] = bhi[i]; }
      }
#pragma unroll
      for (int mi = 0; mi < 2; mi++)
#pragma unroll
        for (int ni = 0; ni < 2; ni++)
          acc[mi][ni] = __builtin_amdgcn_wmma_f32_16x16x32_f16(
              false, a[mi], false, b[ni], (short)0, acc[mi][ni], false, false);
    }
  }

#pragma unroll
  for (int mi = 0; mi < 2; mi++)
#pragma unroll
    for (int ni = 0; ni < 2; ni++) {
      int ng = bn + wn * 32 + ni * 16 + r;
      if (ng < N) {
        int mbase = bm + wm * 32 + mi * 16 + hi * 8;
#pragma unroll
        for (int v = 0; v < 8; v++) {
          size_t idx = (size_t)(mbase + v) * N + ng;
          float val = acc[mi][ni][v];
          if (resid) val += resid[idx];
          if (C)  C[idx]  = val;
          if (Ch) Ch[idx] = (f16)val;
        }
      }
    }
}

// ---------------------------------------------------------------------------
// conv3x3 as 9 accumulated WMMA GEMMs with spatial shift + zero padding
// ---------------------------------------------------------------------------
__global__ __launch_bounds__(256)
void conv3_gemm_kernel(const f16* __restrict__ catH, const f16* __restrict__ W9,
                       float* __restrict__ C) {
  __shared__ alignas(16) f16 lA[BM][BK];
  __shared__ alignas(16) f16 lB[BN][BK];
  int tid  = threadIdx.x;
  int lane = tid & 31, wave = tid >> 5;
  int wm = wave >> 1, wn = wave & 1;
  int bm = blockIdx.x * BM;
  int bn = blockIdx.y * BN;
  int r  = lane & 15, hi = lane >> 4;
  int k1 = hi ? 8 : 0, k2 = hi ? 24 : 16;

  v8f acc[2][2];
#pragma unroll
  for (int mi = 0; mi < 2; mi++)
#pragma unroll
    for (int ni = 0; ni < 2; ni++)
#pragma unroll
      for (int i = 0; i < 8; i++) acc[mi][ni][i] = 0.0f;

  int rowq[4], colq[4], yq[4], xq[4], bq[4];
#pragma unroll
  for (int q = 0; q < 4; q++) {
    int ci = q * 256 + tid;
    rowq[q] = ci >> 3;
    colq[q] = (ci & 7) * 8;
    int m = bm + rowq[q];
    bq[q] = m >> 12;
    int rem = m & 4095;
    yq[q] = rem >> 6;
    xq[q] = rem & 63;
  }

  for (int o = 0; o < 9; o++) {
    int dy = o / 3 - 1, dx = o % 3 - 1;
    size_t ap[4];
    bool vq[4];
#pragma unroll
    for (int q = 0; q < 4; q++) {
      int ys = yq[q] + dy, xs = xq[q] + dx;
      vq[q] = (ys >= 0) && (ys < HH) && (xs >= 0) && (xs < WW);
      ap[q] = ((size_t)(bq[q] << 12) + (size_t)(vq[q] ? (ys << 6) + xs : 0)) * 512;
    }
    const f16* Wo = W9 + (size_t)o * 128 * 512;
    for (int kb = 0; kb < 512; kb += BK) {
      __syncthreads();
#pragma unroll
      for (int q = 0; q < 4; q++) {
        if (vq[q]) {
          cp16_g2l(&catH[ap[q] + kb + colq[q]], &lA[rowq[q]][colq[q]]);
        } else {
          v8h zz;
#pragma unroll
          for (int i = 0; i < 8; i++) zz[i] = (f16)0.0f;
          *(v8h*)&lA[rowq[q]][colq[q]] = zz;
        }
      }
#pragma unroll
      for (int q = 0; q < 2; q++) {
        int ci = q * 256 + tid;
        int row = ci >> 3, col = (ci & 7) * 8;
        cp16_g2l(&Wo[(size_t)(bn + row) * 512 + kb + col], &lB[row][col]);
      }
      wait_async_copies();
      __syncthreads();
#pragma unroll
      for (int ks = 0; ks < BK; ks += 32) {
        v16h a[2], b[2];
#pragma unroll
        for (int mi = 0; mi < 2; mi++) {
          int row = wm * 32 + mi * 16 + r;
          v8h alo = *(const v8h*)&lA[row][ks + k1];
          v8h ahi = *(const v8h*)&lA[row][ks + k2];
#pragma unroll
          for (int i = 0; i < 8; i++) { a[mi][i] = alo[i]; a[mi][i + 8] = ahi[i]; }
        }
#pragma unroll
        for (int ni = 0; ni < 2; ni++) {
          int row = wn * 32 + ni * 16 + r;
          v8h blo = *(const v8h*)&lB[row][ks + k1];
          v8h bhi = *(const v8h*)&lB[row][ks + k2];
#pragma unroll
          for (int i = 0; i < 8; i++) { b[ni][i] = blo[i]; b[ni][i + 8] = bhi[i]; }
        }
#pragma unroll
        for (int mi = 0; mi < 2; mi++)
#pragma unroll
          for (int ni = 0; ni < 2; ni++)
            acc[mi][ni] = __builtin_amdgcn_wmma_f32_16x16x32_f16(
                false, a[mi], false, b[ni], (short)0, acc[mi][ni], false, false);
      }
    }
  }
#pragma unroll
  for (int mi = 0; mi < 2; mi++)
#pragma unroll
    for (int ni = 0; ni < 2; ni++) {
      int ng = bn + wn * 32 + ni * 16 + r;
      int mbase = bm + wm * 32 + mi * 16 + hi * 8;
#pragma unroll
      for (int v = 0; v < 8; v++)
        C[(size_t)(mbase + v) * 128 + ng] = acc[mi][ni][v];
    }
}

// ---------------------------------------------------------------------------
// BN + ReLU + channel-concat -> f16 (M x 256)
// ---------------------------------------------------------------------------
__global__ void bnrelu_kernel(const float* __restrict__ c3, const float* __restrict__ c1,
                              const float* g3, const float* b3, const float* m3, const float* v3,
                              const float* g1, const float* b1, const float* m1, const float* v1,
                              f16* __restrict__ zh) {
  int i = blockIdx.x * blockDim.x + threadIdx.x;
  if (i >= MROWS * 256) return;
  int m = i >> 8, c = i & 255;
  float val;
  if (c < 128) {
    float xx = c3[(size_t)m * 128 + c];
    val = (xx - m3[c]) * rsqrtf(v3[c] + 1e-5f) * g3[c] + b3[c];
  } else {
    int cc = c - 128;
    float xx = c1[(size_t)m * 128 + cc];
    val = (xx - m1[cc]) * rsqrtf(v1[cc] + 1e-5f) * g1[cc] + b1[cc];
  }
  zh[i] = (f16)fmaxf(val, 0.0f);
}

// ---------------------------------------------------------------------------
// LayerNorm over 256 cols; optional f32 and/or f16 output (in-place safe)
// ---------------------------------------------------------------------------
__global__ __launch_bounds__(256)
void ln256_kernel(const float* __restrict__ X, const float* __restrict__ g,
                  const float* __restrict__ bb, float* __restrict__ Y,
                  f16* __restrict__ Yh) {
  int m = blockIdx.x;
  int c = threadIdx.x;
  __shared__ float red[256];
  float x = X[(size_t)m * 256 + c];
  red[c] = x;
  __syncthreads();
  for (int s = 128; s > 0; s >>= 1) { if (c < s) red[c] += red[c + s]; __syncthreads(); }
  float mu = red[0] * (1.0f / 256.0f);
  __syncthreads();
  float d = x - mu;
  red[c] = d * d;
  __syncthreads();
  for (int s = 128; s > 0; s >>= 1) { if (c < s) red[c] += red[c + s]; __syncthreads(); }
  float yv = d * rsqrtf(red[0] * (1.0f / 256.0f) + 1e-5f) * g[c] + bb[c];
  if (Y)  Y[(size_t)m * 256 + c]  = yv;
  if (Yh) Yh[(size_t)m * 256 + c] = (f16)yv;
}

// ---------------------------------------------------------------------------
// sequence permutation map (gather==scatter index)
// ---------------------------------------------------------------------------
__device__ __forceinline__ int seq_map(int br, int j) {
  if (br == 0) return j;
  if (br == 1) { int xx = j >> 6, yy = j & 63; return (yy << 6) + xx; }
  if (br == 2) return (LL - 1) - j;
  int jp = (LL - 1) - j; int xx = jp >> 6, yy = jp & 63; return (yy << 6) + xx;
}

__global__ void gather_kernel(const float* __restrict__ zf, f16* __restrict__ seqh, int br) {
  int i = blockIdx.x * blockDim.x + threadIdx.x;
  if (i >= MROWS * 256) return;
  int c = i & 255, t = (i >> 8) & 4095, b = i >> 20;
  int p = seq_map(br, t);
  seqh[i] = (f16)zf[(((size_t)b << 12) + p) * 256 + c];
}

__global__ void scatter_kernel(const float* __restrict__ bo, float* __restrict__ acc, int br) {
  int i = blockIdx.x * blockDim.x + threadIdx.x;
  if (i >= MROWS * 256) return;
  int c = i & 255, t = (i >> 8) & 4095, b = i >> 20;
  size_t dst = (((size_t)b << 12) + seq_map(br, t)) * 256 + c;
  float v = bo[i];
  if (br == 0) acc[dst] = v; else acc[dst] += v;
}

// ---------------------------------------------------------------------------
// depthwise causal conv1d (k=4) + bias + SiLU -> xBC
// ---------------------------------------------------------------------------
__global__ void dwconv_silu_kernel(const float* __restrict__ zx, const float* __restrict__ cw,
                                   const float* __restrict__ cb, float* __restrict__ xbc) {
  int i = blockIdx.x * blockDim.x + threadIdx.x;
  if (i >= MROWS * CVD) return;
  int ch = i % CVD;
  int mt = i / CVD;
  int t = mt & 4095, b = mt >> 12;
  float a = cb[ch];
#pragma unroll
  for (int k = 0; k < 4; k++) {
    int tt = t - 3 + k;
    if (tt >= 0)
      a += cw[ch * 4 + k] * zx[((size_t)((b << 12) + tt)) * DPJ + DIN + ch];
  }
  xbc[i] = a / (1.0f + expf(-a));
}

__global__ void dtda_kernel(const float* __restrict__ zx, const float* __restrict__ dtb,
                            const float* __restrict__ alog, float* __restrict__ dt,
                            float* __restrict__ dA) {
  int i = blockIdx.x * blockDim.x + threadIdx.x;
  if (i >= MROWS * NHD) return;
  int hd = i & 7, m = i >> 3;
  float v = zx[(size_t)m * DPJ + (DIN + CVD) + hd] + dtb[hd];
  float sp = (v > 20.0f) ? v : log1pf(expf(v));
  dt[i] = sp;
  dA[i] = expf(-expf(alog[hd]) * sp);
}

// ---------------------------------------------------------------------------
// Segmented SSM scan (dA scalar per head => segmentable recurrence)
// ---------------------------------------------------------------------------
__global__ __launch_bounds__(256)
void scan_local_kernel(const float* __restrict__ xbc, const float* __restrict__ dt,
                       const float* __restrict__ dA, float* __restrict__ y,
                       float* __restrict__ cparr, float* __restrict__ hseg,
                       float* __restrict__ Pseg) {
  int seg = blockIdx.x & (SEG - 1);
  int hd  = (blockIdx.x >> 6) & 7;
  int b   = blockIdx.x >> 9;
  int bh  = b * NHD + hd;
  int tid = threadIdx.x;
  int p = tid & 63, nb = (tid >> 6) * 32;
  __shared__ float sx[64], sB[128], sC[128], sr[256];
  float h[32];
#pragma unroll
  for (int i = 0; i < 32; i++) h[i] = 0.0f;
  float cp = 1.0f;
  int t0 = seg * TS;

  for (int tt = 0; tt < TS; tt++) {
    int t = t0 + tt;
    size_t m = (size_t)((b << 12) + t);
    const float* row = xbc + m * CVD;
    if (tid < 64) {
      sx[tid] = row[hd * 64 + tid];
    } else if (tid < 192) {
      sB[tid - 64] = row[DIN + (tid - 64)];
    } else {
      sC[tid - 192] = row[DIN + DST + (tid - 192)];
      sC[tid - 128] = row[DIN + DST + (tid - 128)];
    }
    float dtv = dt[m * NHD + hd];
    float dAv = dA[m * NHD + hd];
    __syncthreads();
    float dtx = dtv * sx[p];
    float part = 0.0f;
#pragma unroll
    for (int i = 0; i < 32; i++) {
      h[i] = h[i] * dAv + dtx * sB[nb + i];
      part += h[i] * sC[nb + i];
    }
    cp *= dAv;
    sr[tid] = part;
    __syncthreads();
    if (tid < 64)
      y[m * DIN + hd * 64 + tid] = sr[tid] + sr[tid + 64] + sr[tid + 128] + sr[tid + 192];
    if (tid == 0) cparr[((size_t)bh << 12) + t] = cp;
    __syncthreads();
  }
  size_t hb = (size_t)(bh * SEG + seg) * 8192 + (size_t)tid * 32;
#pragma unroll
  for (int i = 0; i < 32; i++) hseg[hb + i] = h[i];
  if (tid == 0) Pseg[bh * SEG + seg] = cp;
}

__global__ __launch_bounds__(256)
void scan_prop_kernel(const float* __restrict__ hseg, const float* __restrict__ Pseg,
                      float* __restrict__ hstart) {
  int bh  = blockIdx.x;
  int tid = threadIdx.x;
  float hs[32];
#pragma unroll
  for (int i = 0; i < 32; i++) hs[i] = 0.0f;
  for (int s = 0; s < SEG; s++) {
    size_t base = (size_t)(bh * SEG + s) * 8192 + (size_t)tid * 32;
    float P = Pseg[bh * SEG + s];
#pragma unroll
    for (int i = 0; i < 32; i++) {
      hstart[base + i] = hs[i];
      hs[i] = P * hs[i] + hseg[base + i];
    }
  }
}

__global__ __launch_bounds__(256)
void scan_fix_kernel(const float* __restrict__ xbc, const float* __restrict__ cparr,
                     const float* __restrict__ hstart, float* __restrict__ y) {
  int seg = blockIdx.x & (SEG - 1);
  if (seg == 0) return;                        // h_start == 0
  int hd  = (blockIdx.x >> 6) & 7;
  int b   = blockIdx.x >> 9;
  int bh  = b * NHD + hd;
  int tid = threadIdx.x;
  int p = tid & 63, nb = (tid >> 6) * 32;
  __shared__ float sC[128], sr[256];
  float hs[32];
  size_t base = (size_t)(bh * SEG + seg) * 8192 + (size_t)tid * 32;
#pragma unroll
  for (int i = 0; i < 32; i++) hs[i] = hstart[base + i];
  int t0 = seg * TS;
  for (int tt = 0; tt < TS; tt++) {
    int t = t0 + tt;
    size_t m = (size_t)((b << 12) + t);
    if (tid < 128) sC[tid] = xbc[m * CVD + DIN + DST + tid];
    __syncthreads();
    float part = 0.0f;
#pragma unroll
    for (int i = 0; i < 32; i++) part += hs[i] * sC[nb + i];
    sr[tid] = part;
    __syncthreads();
    if (tid < 64)
      y[m * DIN + hd * 64 + tid] +=
          cparr[((size_t)bh << 12) + t] * (sr[tid] + sr[tid + 64] + sr[tid + 128] + sr[tid + 192]);
    __syncthreads();
  }
}

// ---------------------------------------------------------------------------
// y = (y_ssm + D*xh) * silu(z); RMSNorm(512)*norm_g -> f16
// ---------------------------------------------------------------------------
__global__ __launch_bounds__(256)
void gate_rms_kernel(const float* __restrict__ y, const float* __restrict__ xbc,
                     const float* __restrict__ zx, const float* __restrict__ Dp,
                     const float* __restrict__ ng, f16* __restrict__ yh) {
  int m = blockIdx.x;
  int tid = threadIdx.x;
  __shared__ float red[256];
  float v0, v1, ss = 0.0f;
  {
    int c = tid;
    float xh = xbc[(size_t)m * CVD + c];
    float z  = zx[(size_t)m * DPJ + c];
    float yy = y[(size_t)m * DIN + c] + Dp[c >> 6] * xh;
    v0 = yy * (z / (1.0f + expf(-z)));
    ss += v0 * v0;
  }
  {
    int c = tid + 256;
    float xh = xbc[(size_t)m * CVD + c];
    float z  = zx[(size_t)m * DPJ + c];
    float yy = y[(size_t)m * DIN + c] + Dp[c >> 6] * xh;
    v1 = yy * (z / (1.0f + expf(-z)));
    ss += v1 * v1;
  }
  red[tid] = ss;
  __syncthreads();
  for (int s = 128; s > 0; s >>= 1) { if (tid < s) red[tid] += red[tid + s]; __syncthreads(); }
  float inv = rsqrtf(red[0] * (1.0f / 512.0f) + 1e-5f);
  yh[(size_t)m * DIN + tid]       = (f16)(v0 * inv * ng[tid]);
  yh[(size_t)m * DIN + tid + 256] = (f16)(v1 * inv * ng[tid + 256]);
}

__global__ void final_kernel(const float* __restrict__ curr, const float* __restrict__ acc,
                             float* __restrict__ out) {
  int i = blockIdx.x * blockDim.x + threadIdx.x;
  if (i >= MROWS * 256) return;
  out[i] = curr[i] + 0.25f * acc[i];
}

// ---------------------------------------------------------------------------
// host orchestration
// ---------------------------------------------------------------------------
extern "C" void kernel_launch(void* const* d_in, const int* in_sizes, int n_in,
                              void* d_out, int out_size, void* d_ws, size_t ws_size,
                              hipStream_t stream) {
  const float* curr    = (const float*)d_in[0];
  const float* prev    = (const float*)d_in[1];
  const float* conv3_w = (const float*)d_in[2];
  const float* bn3_g   = (const float*)d_in[3];
  const float* bn3_b   = (const float*)d_in[4];
  const float* bn3_m   = (const float*)d_in[5];
  const float* bn3_v   = (const float*)d_in[6];
  const float* conv1_w = (const float*)d_in[7];
  const float* bn1_g   = (const float*)d_in[8];
  const float* bn1_b   = (const float*)d_in[9];
  const float* bn1_m   = (const float*)d_in[10];
  const float* bn1_v   = (const float*)d_in[11];
  const float* fuse_W  = (const float*)d_in[12];
  const float* fuse_b  = (const float*)d_in[13];
  const float* ln_g    = (const float*)d_in[14];
  const float* ln_b    = (const float*)d_in[15];
  const float* inp_W   = (const float*)d_in[16];
  const float* inp_b   = (const float*)d_in[17];
  const float* mln_g   = (const float*)d_in[18];
  const float* mln_b   = (const float*)d_in[19];
  const float* min_W   = (const float*)d_in[20];
  const float* min_b   = (const float*)d_in[21];
  const float* conv_w  = (const float*)d_in[22];
  const float* conv_b  = (const float*)d_in[23];
  const float* A_log   = (const float*)d_in[24];
  const float* Dp      = (const float*)d_in[25];
  const float* dt_bias = (const float*)d_in[26];
  const float* mnorm_g = (const float*)d_in[27];
  const float* mout_W  = (const float*)d_in[28];
  const float* mout_b  = (const float*)d_in[29];
  const float* outp_W  = (const float*)d_in[30];
  const float* outp_b  = (const float*)d_in[31];
  float* out = (float*)d_out;

  char* base = (char*)d_ws;
  size_t off = 0;
  auto alloc = [&](size_t bytes) -> char* {
    char* p = base + off;
    off = (off + bytes + 255) & ~(size_t)255;
    return p;
  };

  // f16 weights
  f16* w_fuse  = (f16*)alloc((size_t)256 * 256 * 2);
  f16* w_conv1 = (f16*)alloc((size_t)128 * 512 * 2);
  f16* w_conv3 = (f16*)alloc((size_t)9 * 128 * 512 * 2);
  f16* w_inp   = (f16*)alloc((size_t)4 * 256 * 256 * 2);
  f16* w_min   = (f16*)alloc((size_t)4 * DPJ * 256 * 2);
  f16* w_mout  = (f16*)alloc((size_t)4 * 256 * 512 * 2);
  f16* w_outp  = (f16*)alloc((size_t)4 * 256 * 256 * 2);
  // activations
  f16*   cat_h = (f16*)alloc((size_t)MROWS * 512 * 2);
  float* c3    = (float*)alloc((size_t)MROWS * 128 * 4);
  float* c1    = (float*)alloc((size_t)MROWS * 128 * 4);
  f16*   z_h   = (f16*)alloc((size_t)MROWS * 256 * 2);
  float* zf    = (float*)alloc((size_t)MROWS * 256 * 4);
  f16*   seq_h = (f16*)alloc((size_t)MROWS * 256 * 2);
  float* s     = (float*)alloc((size_t)MROWS * 256 * 4);
  f16*   x_h   = (f16*)alloc((size_t)MROWS * 256 * 2);
  float* zx    = (float*)alloc((size_t)MROWS * DPJ * 4);
  float* xbc   = (float*)alloc((size_t)MROWS * CVD * 4);
  float* dt    = (float*)alloc((size_t)MROWS * NHD * 4);
  float* dA    = (float*)alloc((size_t)MROWS * NHD * 4);
  float* y_ssm = (float*)alloc((size_t)MROWS * DIN * 4);
  f16*   yn_h  = (f16*)alloc((size_t)MROWS * DIN * 2);
  f16*   t2_h  = (f16*)alloc((size_t)MROWS * 256 * 2);
  float* bo    = (float*)alloc((size_t)MROWS * 256 * 4);
  float* accb  = (float*)alloc((size_t)MROWS * 256 * 4);
  // scan scratch
  float* cparr  = (float*)alloc((size_t)32 * LL * 4);
  float* hseg   = (float*)alloc((size_t)32 * SEG * 8192 * 4);
  float* hstart = (float*)alloc((size_t)32 * SEG * 8192 * 4);
  float* Pseg   = (float*)alloc((size_t)32 * SEG * 4);

  auto blocks = [](long n) { return (unsigned)((n + 255) / 256); };

  // weight conversion
  f2h_kernel<<<blocks(256 * 256), 256, 0, stream>>>(fuse_W, w_fuse, 256 * 256);
  f2h_kernel<<<blocks(128 * 512), 256, 0, stream>>>(conv1_w, w_conv1, 128 * 512);
  conv3_repack_kernel<<<blocks(9 * 128 * 512), 256, 0, stream>>>(conv3_w, w_conv3);
  f2h_kernel<<<blocks(4 * 256 * 256), 256, 0, stream>>>(inp_W, w_inp, 4 * 256 * 256);
  f2h_kernel<<<blocks(4 * DPJ * 256), 256, 0, stream>>>(min_W, w_min, 4 * DPJ * 256);
  f2h_kernel<<<blocks(4 * 256 * 512), 256, 0, stream>>>(mout_W, w_mout, 4 * 256 * 512);
  f2h_kernel<<<blocks(4 * 256 * 256), 256, 0, stream>>>(outp_W, w_outp, 4 * 256 * 256);
  build_cat_kernel<<<blocks((long)MROWS * 512), 256, 0, stream>>>(curr, prev, cat_h);

  // fusion stage
  conv3_gemm_kernel<<<dim3(MROWS / BM, 2), 256, 0, stream>>>(cat_h, w_conv3, c3);
  gemm_wmma_kernel<<<dim3(MROWS / BM, 2), 256, 0, stream>>>(cat_h, w_conv1, nullptr, nullptr,
                                                            c1, nullptr, MROWS, 128, 512);
  bnrelu_kernel<<<blocks((long)MROWS * 256), 256, 0, stream>>>(
      c3, c1, bn3_g, bn3_b, bn3_m, bn3_v, bn1_g, bn1_b, bn1_m, bn1_v, z_h);
  gemm_wmma_kernel<<<dim3(MROWS / BM, 4), 256, 0, stream>>>(z_h, w_fuse, fuse_b, nullptr,
                                                            zf, nullptr, MROWS, 256, 256);
  ln256_kernel<<<MROWS, 256, 0, stream>>>(zf, ln_g, ln_b, zf, nullptr);

  // branches (sequential, buffers reused)
  for (int br = 0; br < 4; br++) {
    gather_kernel<<<blocks((long)MROWS * 256), 256, 0, stream>>>(zf, seq_h, br);
    gemm_wmma_kernel<<<dim3(MROWS / BM, 4), 256, 0, stream>>>(
        seq_h, w_inp + (size_t)br * 256 * 256, inp_b + br * 256, nullptr,
        s, nullptr, MROWS, 256, 256);
    ln256_kernel<<<MROWS, 256, 0, stream>>>(s, mln_g + br * 256, mln_b + br * 256,
                                            nullptr, x_h);
    gemm_wmma_kernel<<<dim3(MROWS / BM, (DPJ + BN - 1) / BN), 256, 0, stream>>>(
        x_h, w_min + (size_t)br * DPJ * 256, min_b + br * DPJ, nullptr,
        zx, nullptr, MROWS, DPJ, 256);
    dwconv_silu_kernel<<<blocks((long)MROWS * CVD), 256, 0, stream>>>(
        zx, conv_w + (size_t)br * CVD * 4, conv_b + br * CVD, xbc);
    dtda_kernel<<<blocks((long)MROWS * NHD), 256, 0, stream>>>(
        zx, dt_bias + br * NHD, A_log + br * NHD, dt, dA);
    scan_local_kernel<<<BATCH * NHD * SEG, 256, 0, stream>>>(xbc, dt, dA, y_ssm,
                                                             cparr, hseg, Pseg);
    scan_prop_kernel<<<BATCH * NHD, 256, 0, stream>>>(hseg, Pseg, hstart);
    scan_fix_kernel<<<BATCH * NHD * SEG, 256, 0, stream>>>(xbc, cparr, hstart, y_ssm);
    gate_rms_kernel<<<MROWS, 256, 0, stream>>>(y_ssm, xbc, zx, Dp + br * NHD,
                                               mnorm_g + br * DIN, yn_h);
    gemm_wmma_kernel<<<dim3(MROWS / BM, 4), 256, 0, stream>>>(
        yn_h, w_mout + (size_t)br * 256 * 512, mout_b + br * 256, /*resid=*/s,
        nullptr, t2_h, MROWS, 256, 512);
    gemm_wmma_kernel<<<dim3(MROWS / BM, 4), 256, 0, stream>>>(
        t2_h, w_outp + (size_t)br * 256 * 256, outp_b + br * 256, nullptr,
        bo, nullptr, MROWS, 256, 256);
    scatter_kernel<<<blocks((long)MROWS * 256), 256, 0, stream>>>(bo, accb, br);
  }

  final_kernel<<<blocks((long)MROWS * 256), 256, 0, stream>>>(curr, accb, out);
}
